// GAT_74036646248596
// MI455X (gfx1250) — matrix operand order repaired and checked by reading, not compiled, not run on hardware
//
#include <hip/hip_runtime.h>
#include <hip/hip_bf16.h>
#include <math.h>

// ---------------------------------------------------------------------------
// GAT (2x GATConv + mean-pool + linear) for MI455X / gfx1250, wave32, WMMA.
//   N=20000 nodes, E=640000 edges, IND=512, HID=128, HID2=64, OUT=2, G=16
// GEMMs in bf16 via v_wmma_f32_16x16x32_bf16 with transposed-B storage
// (contiguous 32-bit fragment loads on both operands) and 4x (layer 1) / 2x (layer 2) M-subtile
// register blocking to amortize B loads.  Edge softmax/aggregation via
// float atomics (atomic-bound, degree ~33); score dots via wave32 shuffles.
// ---------------------------------------------------------------------------

#define IND  512
#define HID  128
#define HID2 64
#define OUTD 2
#define NGR  16
#define NEG_SLOPE 0.2f

typedef __attribute__((ext_vector_type(16))) __bf16 v16bf;
typedef __attribute__((ext_vector_type(8)))  float  v8f;

__device__ __forceinline__ unsigned short f2bf(float f) {
    unsigned int u = __float_as_uint(f);
    // round-to-nearest-even
    unsigned int r = (u + 0x7FFFu + ((u >> 16) & 1u)) >> 16;
    return (unsigned short)r;
}

__device__ __forceinline__ void atomicMaxF(float* addr, float v) {
    // works when *addr initialized to -inf: positive path via signed max,
    // negative path via unsigned min (bit patterns of negatives grow downward)
    if (v >= 0.0f) atomicMax((int*)addr, __float_as_int(v));
    else           atomicMin((unsigned int*)addr, __float_as_uint(v));
}

// ---------------------------------------------------------------------------
// fp32 -> bf16 conversion (row-major preserved)
// ---------------------------------------------------------------------------
__global__ void k_cvt_bf16(const float* __restrict__ in,
                           unsigned short* __restrict__ out, int n) {
    int i = blockIdx.x * blockDim.x + threadIdx.x;
    if (i < n) out[i] = f2bf(in[i]);
}

// fp32 [K x Ncols] -> bf16 transposed [Ncols x K]  (for WMMA B operand)
__global__ void k_cvt_bf16_tr(const float* __restrict__ in,
                              unsigned short* __restrict__ out,
                              int K_, int Ncols) {
    int i = blockIdx.x * blockDim.x + threadIdx.x;
    if (i >= K_ * Ncols) return;
    int k = i / Ncols, n2 = i - k * Ncols;     // contiguous reads
    out[(size_t)n2 * K_ + k] = f2bf(in[i]);
}

// ---------------------------------------------------------------------------
// WMMA GEMM:  C[M x NTOT] = A[M x K] * B[K x NTOT]
//   A  bf16 row-major [M x K],  BT bf16 transposed [NTOT x K],  C fp32.
// One wave computes a (16*MREP) x 16 output block; blockDim = (32, NTOT/16).
// Register fragment layouts used here (wave32; contiguous k order, same on A and B):
//   A 16x32:  lane = {half,M%16}; vgpr i -> K = 16*half + 2i
//   B 32x16:  lane = {half,N%16}; vgpr i -> K = 16*half + 2i
//   C 16x16:  vgpr r -> M = r + 8*half, N = lane%16
// With BT storage both fragments load as 8 contiguous b32 per K-step.
// ---------------------------------------------------------------------------
template <int K, int NTOT, int MREP>
__global__ void k_wmma_gemm_bf16(const unsigned short* __restrict__ A,
                                 const unsigned short* __restrict__ BT,
                                 float* __restrict__ C, int M) {
    const int lane = threadIdx.x;          // 0..31
    const int half = lane >> 4;            // 0/1
    const int l16  = lane & 15;
    const int tileN = threadIdx.y;
    const int n = tileN * 16 + l16;
    const int mBase = blockIdx.x * (16 * MREP);

    const unsigned short* Brow = BT + (size_t)n * K;

    const unsigned short* Arow[MREP];
#pragma unroll
    for (int r = 0; r < MREP; ++r) {
        int row = mBase + r * 16 + l16;
        if (row >= M) row = M - 1;         // clamp (stores guarded below)
        Arow[r] = A + (size_t)row * K;
    }

    v8f acc[MREP];
#pragma unroll
    for (int r = 0; r < MREP; ++r) acc[r] = (v8f){};

    for (int k0 = 0; k0 < K; k0 += 32) {
        union { v16bf v; unsigned int u[8]; } b;
#pragma unroll
        for (int i = 0; i < 8; ++i)        // K = k0 + 16*half + 2i, 2i+1
            b.u[i] = *(const unsigned int*)(Brow + k0 + half * 16 + 2 * i);
#pragma unroll
        for (int r = 0; r < MREP; ++r) {
            union { v16bf v; unsigned int u[8]; } a;
#pragma unroll
            for (int i = 0; i < 8; ++i) {
                const int ka = k0 + half * 16 + 2 * i;
                a.u[i] = *(const unsigned int*)(Arow[r] + ka);
            }
            acc[r] = __builtin_amdgcn_wmma_f32_16x16x32_bf16(
                /*neg_a=*/false, a.v, /*neg_b=*/false, b.v,
                /*c_mod=*/(short)0, acc[r], /*reuse_a=*/false, /*reuse_b=*/false);
        }
    }
#pragma unroll
    for (int r = 0; r < MREP; ++r) {
#pragma unroll
        for (int q = 0; q < 8; ++q) {
            const int m = mBase + r * 16 + q + 8 * half;
            if (m < M) C[(size_t)m * NTOT + tileN * 16 + l16] = acc[r][q];
        }
    }
}

// ---------------------------------------------------------------------------
// Per-node attention scores (h . a_src, h . a_dst) + init of per-layer
// accumulators.  One wave32 per node.
// ---------------------------------------------------------------------------
template <int F>
__global__ void k_scores_init(const float* __restrict__ h,
                              const float* __restrict__ a_src,
                              const float* __restrict__ a_dst,
                              float* __restrict__ ss, float* __restrict__ sd,
                              float* __restrict__ maxv,
                              float* __restrict__ denom,
                              float* __restrict__ agg, int n) {
    const int wave = (blockIdx.x * blockDim.x + threadIdx.x) >> 5;
    const int lane = threadIdx.x & 31;
    if (wave >= n) return;
    const float* hp = h + (size_t)wave * F;
    float* ap = agg + (size_t)wave * F;
    float s0 = 0.f, s1 = 0.f;
#pragma unroll
    for (int f = lane; f < F; f += 32) {
        const float v = hp[f];
        s0 += v * a_src[f];
        s1 += v * a_dst[f];
        ap[f] = 0.0f;
    }
#pragma unroll
    for (int off = 16; off > 0; off >>= 1) {
        s0 += __shfl_xor(s0, off, 32);
        s1 += __shfl_xor(s1, off, 32);
    }
    if (lane == 0) {
        ss[wave] = s0;
        sd[wave] = s1;
        maxv[wave] = -INFINITY;
        denom[wave] = 0.0f;
    }
}

// ---------------------------------------------------------------------------
// Edge pass 1: e = leaky_relu(ss[s] + sd[d]);  segment max over dst.
// Indices t in [0, E) are real edges, t in [E, E+N) are self loops.
// ---------------------------------------------------------------------------
__global__ void k_edge_scores(const int* __restrict__ src,
                              const int* __restrict__ dst,
                              const float* __restrict__ ss,
                              const float* __restrict__ sd,
                              float* __restrict__ ev,
                              float* __restrict__ maxv, int E, int T) {
    const int t = blockIdx.x * blockDim.x + threadIdx.x;
    if (t >= T) return;
    const int s = (t < E) ? src[t] : (t - E);
    const int d = (t < E) ? dst[t] : (t - E);
    float e = ss[s] + sd[d];
    e = (e > 0.0f) ? e : NEG_SLOPE * e;
    ev[t] = e;
    atomicMaxF(maxv + d, e);
}

// Edge pass 2: ex = exp(e - max[d]); segment sum over dst.
__global__ void k_edge_exp(const int* __restrict__ dst,
                           float* __restrict__ ev,
                           const float* __restrict__ maxv,
                           float* __restrict__ denom, int E, int T) {
    const int t = blockIdx.x * blockDim.x + threadIdx.x;
    if (t >= T) return;
    const int d = (t < E) ? dst[t] : (t - E);
    const float ex = __expf(ev[t] - maxv[d]);
    ev[t] = ex;
    atomicAdd(denom + d, ex);
}

// Edge pass 3: agg[d] += (ex/denom[d]) * h[s].  One wave32 per edge.
template <int F>
__global__ void k_edge_agg(const int* __restrict__ src,
                           const int* __restrict__ dst,
                           const float* __restrict__ ev,
                           const float* __restrict__ denom,
                           const float* __restrict__ h,
                           float* __restrict__ agg, int E, int T) {
    const int w = (blockIdx.x * blockDim.x + threadIdx.x) >> 5;
    const int lane = threadIdx.x & 31;
    if (w >= T) return;
    const int s = (w < E) ? src[w] : (w - E);
    const int d = (w < E) ? dst[w] : (w - E);
    const float alpha = ev[w] / denom[d];
    const float* hs = h + (size_t)s * F;
    float* ad = agg + (size_t)d * F;
#pragma unroll
    for (int f = lane; f < F; f += 32) atomicAdd(ad + f, alpha * hs[f]);
}

// ---------------------------------------------------------------------------
// bias + relu epilogues
// ---------------------------------------------------------------------------
template <int F>
__global__ void k_bias_relu_bf16(const float* __restrict__ agg,
                                 const float* __restrict__ bias,
                                 unsigned short* __restrict__ out, int n) {
    const int i = blockIdx.x * blockDim.x + threadIdx.x;
    if (i >= n * F) return;
    float v = agg[i] + bias[i & (F - 1)];
    v = (v > 0.0f) ? v : 0.0f;
    out[i] = f2bf(v);
}

template <int F>
__global__ void k_bias_relu_f32(const float* __restrict__ agg,
                                const float* __restrict__ bias,
                                float* __restrict__ out, int n) {
    const int i = blockIdx.x * blockDim.x + threadIdx.x;
    if (i >= n * F) return;
    float v = agg[i] + bias[i & (F - 1)];
    out[i] = (v > 0.0f) ? v : 0.0f;
}

// ---------------------------------------------------------------------------
// mean pool over batch + final linear
// ---------------------------------------------------------------------------
__global__ void k_pool_init(float* __restrict__ pooled,
                            float* __restrict__ counts) {
    for (int i = threadIdx.x; i < NGR * HID2; i += blockDim.x) pooled[i] = 0.f;
    if (threadIdx.x < NGR) counts[threadIdx.x] = 0.f;
}

__global__ void k_pool(const float* __restrict__ r2,
                       const int* __restrict__ batch,
                       float* __restrict__ pooled,
                       float* __restrict__ counts, int n) {
    const int wave = (blockIdx.x * blockDim.x + threadIdx.x) >> 5;
    const int lane = threadIdx.x & 31;
    if (wave >= n) return;
    const int g = batch[wave];
    const float* rp = r2 + (size_t)wave * HID2;
#pragma unroll
    for (int f = lane; f < HID2; f += 32) atomicAdd(pooled + g * HID2 + f, rp[f]);
    if (lane == 0) atomicAdd(counts + g, 1.0f);
}

__global__ void k_final(const float* __restrict__ pooled,
                        const float* __restrict__ counts,
                        const float* __restrict__ Wl,
                        const float* __restrict__ bl,
                        float* __restrict__ out) {
    const int t = threadIdx.x;
    if (t >= NGR * OUTD) return;
    const int g = t / OUTD, o = t % OUTD;
    float c = counts[g];
    c = (c > 1.0f) ? c : 1.0f;
    float s = 0.0f;
#pragma unroll
    for (int k = 0; k < HID2; ++k)
        s += (pooled[g * HID2 + k] / c) * Wl[k * OUTD + o];
    out[t] = s + bl[o];
}

// ---------------------------------------------------------------------------
// launcher
// ---------------------------------------------------------------------------
extern "C" void kernel_launch(void* const* d_in, const int* in_sizes, int n_in,
                              void* d_out, int out_size, void* d_ws,
                              size_t ws_size, hipStream_t stream) {
    const float* x     = (const float*)d_in[0];
    const int*   edge  = (const int*)d_in[1];
    const int*   batch = (const int*)d_in[2];
    const float* W1  = (const float*)d_in[3];
    const float* a1s = (const float*)d_in[4];
    const float* a1d = (const float*)d_in[5];
    const float* b1  = (const float*)d_in[6];
    const float* W2  = (const float*)d_in[7];
    const float* a2s = (const float*)d_in[8];
    const float* a2d = (const float*)d_in[9];
    const float* b2  = (const float*)d_in[10];
    const float* Wl  = (const float*)d_in[11];
    const float* bl  = (const float*)d_in[12];

    const int N = in_sizes[2];        // 20000
    const int E = in_sizes[1] / 2;    // 640000
    const int T = E + N;              // edges incl. self-loops
    const int* src = edge;
    const int* dst = edge + E;

    // ---- workspace carving (256B aligned), with buffer reuse -------------
    char* base = (char*)d_ws;
    size_t off = 0;
    auto carve = [&](size_t bytes) -> void* {
        void* r = base + off;
        off = (off + bytes + 255) & ~(size_t)255;
        return r;
    };
    unsigned short* xb   = (unsigned short*)carve((size_t)N * IND * 2);
    unsigned short* w1bt = (unsigned short*)carve((size_t)IND * HID * 2);
    unsigned short* w2bt = (unsigned short*)carve((size_t)HID * HID2 * 2);
    float* hbuf  = (float*)carve((size_t)N * HID * 4);   // h1, then h2
    float* aggb  = (float*)carve((size_t)N * HID * 4);   // agg1, then agg2
    unsigned short* h1b = (unsigned short*)carve((size_t)N * HID * 2);
    float* ss    = (float*)carve((size_t)N * 4);
    float* sd    = (float*)carve((size_t)N * 4);
    float* maxv  = (float*)carve((size_t)N * 4);
    float* denom = (float*)carve((size_t)N * 4);
    float* ev    = (float*)carve((size_t)T * 4);
    float* pooled = (float*)carve((size_t)NGR * HID2 * 4);
    float* counts = (float*)carve((size_t)NGR * 4);
    float* r2 = (float*)xb;           // alias: x-bf16 dead after GEMM1

    const int BT = 256;

    // ---- precision conversion (weights transposed for WMMA B side) ------
    k_cvt_bf16<<<(N * IND + BT - 1) / BT, BT, 0, stream>>>(x, xb, N * IND);
    k_cvt_bf16_tr<<<(IND * HID + BT - 1) / BT, BT, 0, stream>>>(W1, w1bt,
                                                                IND, HID);
    k_cvt_bf16_tr<<<(HID * HID2 + BT - 1) / BT, BT, 0, stream>>>(W2, w2bt,
                                                                 HID, HID2);

    // ---- layer 1 ---------------------------------------------------------
    {
        const int MREP = 4;
        const int grid = (N + 16 * MREP - 1) / (16 * MREP);
        k_wmma_gemm_bf16<IND, HID, MREP>
            <<<dim3(grid), dim3(32, HID / 16), 0, stream>>>(xb, w1bt, hbuf, N);
    }
    k_scores_init<HID><<<(N * 32 + BT - 1) / BT, BT, 0, stream>>>(
        hbuf, a1s, a1d, ss, sd, maxv, denom, aggb, N);
    k_edge_scores<<<(T + BT - 1) / BT, BT, 0, stream>>>(src, dst, ss, sd, ev,
                                                        maxv, E, T);
    k_edge_exp<<<(T + BT - 1) / BT, BT, 0, stream>>>(dst, ev, maxv, denom, E, T);
    k_edge_agg<HID><<<((size_t)T * 32 + BT - 1) / BT, BT, 0, stream>>>(
        src, dst, ev, denom, hbuf, aggb, E, T);
    k_bias_relu_bf16<HID><<<(N * HID + BT - 1) / BT, BT, 0, stream>>>(
        aggb, b1, h1b, N);

    // ---- layer 2 ---------------------------------------------------------
    {
        const int MREP = 2;
        const int grid = (N + 16 * MREP - 1) / (16 * MREP);
        k_wmma_gemm_bf16<HID, HID2, MREP>
            <<<dim3(grid), dim3(32, HID2 / 16), 0, stream>>>(h1b, w2bt, hbuf, N);
    }
    k_scores_init<HID2><<<(N * 32 + BT - 1) / BT, BT, 0, stream>>>(
        hbuf, a2s, a2d, ss, sd, maxv, denom, aggb, N);
    k_edge_scores<<<(T + BT - 1) / BT, BT, 0, stream>>>(src, dst, ss, sd, ev,
                                                        maxv, E, T);
    k_edge_exp<<<(T + BT - 1) / BT, BT, 0, stream>>>(dst, ev, maxv, denom, E, T);
    k_edge_agg<HID2><<<((size_t)T * 32 + BT - 1) / BT, BT, 0, stream>>>(
        src, dst, ev, denom, hbuf, aggb, E, T);
    k_bias_relu_f32<HID2><<<(N * HID2 + BT - 1) / BT, BT, 0, stream>>>(
        aggb, b2, r2, N);

    // ---- pool + classifier ----------------------------------------------
    k_pool_init<<<1, 256, 0, stream>>>(pooled, counts);
    k_pool<<<(N * 32 + BT - 1) / BT, BT, 0, stream>>>(r2, batch, pooled,
                                                      counts, N);
    k_final<<<1, 64, 0, stream>>>(pooled, counts, Wl, bl, (float*)d_out);
}
